// MambaSkipFusion_65360812311158
// MI455X (gfx1250) — matrix operand order repaired and checked by reading, not compiled, not run on hardware
//
#include <hip/hip_runtime.h>
#include <hip/hip_bf16.h>

// ---------------------------------------------------------------------------
// Shapes
// ---------------------------------------------------------------------------
#define Bz   2
#define Lsz  4096            // H*W
#define Msz  (Bz*Lsz)        // 8192 token rows
#define DIMc 192
#define DIc  384
#define Nst  16
#define Rr   12
#define XPN  64              // x_proj N padded 44 -> 64

typedef __bf16 bf16;
typedef bf16  v16bf __attribute__((ext_vector_type(16)));
typedef float v8f   __attribute__((ext_vector_type(8)));

// ---- gfx1250 async global->LDS copy (ASYNCcnt path), with safe fallback ----
#if defined(__has_builtin)
#  if __has_builtin(__builtin_amdgcn_global_load_async_to_lds_b128)
#    define HAVE_ASYNC_LDS 1
#  endif
#endif
#ifndef HAVE_ASYNC_LDS
#  define HAVE_ASYNC_LDS 0
#endif

#if HAVE_ASYNC_LDS
typedef int v4i32 __attribute__((vector_size(16)));
typedef __attribute__((address_space(1))) v4i32* gbl_v4p;   // global int4*
typedef __attribute__((address_space(3))) v4i32* lds_v4p;   // shared int4*
#  if __has_builtin(__builtin_amdgcn_s_wait_asynccnt)
#    define WAIT_ASYNC() __builtin_amdgcn_s_wait_asynccnt(0)
#  else
#    define WAIT_ASYNC() asm volatile("s_wait_asynccnt 0x0" ::: "memory")
#  endif
#  define ASYNC_CP16(gsrc, ldst)                                              \
     __builtin_amdgcn_global_load_async_to_lds_b128(                          \
        (gbl_v4p)(gsrc), (lds_v4p)(ldst), 0, 0)
#endif

__device__ __forceinline__ unsigned short f2bf(float f) {
    unsigned int u = __float_as_uint(f);
    unsigned int r = (u + 0x7FFFu + ((u >> 16) & 1u)) >> 16;
    return (unsigned short)r;
}

// ---------------------------------------------------------------------------
// Pack concat(encoder, decoder) (b,c,l) -> bf16 row-major [8192][384]
// ---------------------------------------------------------------------------
__global__ void pack_fuse_A(const float* __restrict__ enc,
                            const float* __restrict__ dec,
                            unsigned short* __restrict__ A1a) {
    int m = blockIdx.x, k = threadIdx.x;      // blockDim = 384
    int b = m >> 12, l = m & (Lsz - 1);
    float v = (k < DIMc)
        ? enc[((size_t)b * DIMc + k)          * Lsz + l]
        : dec[((size_t)b * DIMc + (k - DIMc)) * Lsz + l];
    A1a[(size_t)m * (2 * DIMc) + k] = f2bf(v);
}

// f32 -> bf16 weight convert, zero-pad tail
__global__ void cvt_w(const float* __restrict__ src, unsigned short* __restrict__ dst,
                      int nsrc, int ndst) {
    int i = blockIdx.x * 256 + threadIdx.x;
    if (i < ndst) dst[i] = (i < nsrc) ? f2bf(src[i]) : (unsigned short)0;
}

// ---------------------------------------------------------------------------
// bf16 WMMA GEMM:  C[M x N] = A[M x K] * W[N x K]^T (+bias)
// A row-major [M][lda], W row-major [N][K], both bf16 with 16B-aligned rows.
// Block tile 128(M) x 64(N), K-step 32; 8 waves, each 32x32 via 4 wmma.
// Double-buffered LDS; staging via gfx1250 async global->LDS DMA when
// available (ASYNCcnt), else vectorized b128 register relay.
// ---------------------------------------------------------------------------
#define BLK_M 128
#define BLK_N 64
#define BLK_K 32
#define LDA   34     // LDS row stride in elements (bank-conflict pad)
#define LDB   34

__launch_bounds__(256)
__global__ void gemm_bf16_kernel(
    const unsigned short* __restrict__ A, int lda,
    const unsigned short* __restrict__ W, int Kdim,
    const float* __restrict__ bias,
    float* __restrict__ Cf, int cm, int cn,
    unsigned short* __restrict__ Cb, int bm, int bn)
{
    __shared__ unsigned short As[2][BLK_M * LDA];
    __shared__ unsigned short Bs[2][BLK_N * LDB];

    const int tid   = threadIdx.x;
    const int lane  = tid & 31;
    const int wave  = tid >> 5;
    const int waveM = wave >> 1;        // 0..3
    const int waveN = wave & 1;         // 0..1
    const int g     = lane >> 4;        // lane half
    const int l16   = lane & 15;
    const int m0    = blockIdx.y * BLK_M;
    const int n0    = blockIdx.x * BLK_N;

    // staging geometry: A = 512 16B vectors (2/thread), B = 256 vectors (1/thread)
    const int aRow0 = tid >> 2;                    // 0..63
    const int aCol0 = (tid & 3) * 8;
    const int aRow1 = aRow0 + 64;                  // 64..127
    const int aCol1 = aCol0;
    const int bRow  = tid >> 2;                    // 0..63
    const int bCol  = (tid & 3) * 8;

    const unsigned short* aPtr0 = A + (size_t)(m0 + aRow0) * lda + aCol0;
    const unsigned short* aPtr1 = A + (size_t)(m0 + aRow1) * lda + aCol1;
    const unsigned short* bPtr  = W + (size_t)(n0 + bRow) * Kdim + bCol;

    unsigned short* lA0[2] = { &As[0][aRow0 * LDA + aCol0], &As[1][aRow0 * LDA + aCol0] };
    unsigned short* lA1[2] = { &As[0][aRow1 * LDA + aCol1], &As[1][aRow1 * LDA + aCol1] };
    unsigned short* lB [2] = { &Bs[0][bRow  * LDB + bCol ], &Bs[1][bRow  * LDB + bCol ] };

    v8f acc[2][2];
    #pragma unroll
    for (int mt = 0; mt < 2; ++mt)
        #pragma unroll
        for (int nt = 0; nt < 2; ++nt)
            #pragma unroll
            for (int r = 0; r < 8; ++r) acc[mt][nt][r] = 0.0f;

    // ---- prologue: stage K-tile 0 into buffer 0 ----
#if HAVE_ASYNC_LDS
    ASYNC_CP16(aPtr0, lA0[0]);
    ASYNC_CP16(aPtr1, lA1[0]);
    ASYNC_CP16(bPtr,  lB[0]);
    WAIT_ASYNC();
#else
    {
        uint4 ra0 = *(const uint4*)(aPtr0);
        uint4 ra1 = *(const uint4*)(aPtr1);
        uint4 rb  = *(const uint4*)(bPtr);
        unsigned int* dp;
        dp = (unsigned int*)lA0[0]; dp[0] = ra0.x; dp[1] = ra0.y; dp[2] = ra0.z; dp[3] = ra0.w;
        dp = (unsigned int*)lA1[0]; dp[0] = ra1.x; dp[1] = ra1.y; dp[2] = ra1.z; dp[3] = ra1.w;
        dp = (unsigned int*)lB[0];  dp[0] = rb.x;  dp[1] = rb.y;  dp[2] = rb.z;  dp[3] = rb.w;
    }
#endif
    __syncthreads();

    int cur = 0;
    for (int k0 = 0; k0 < Kdim; k0 += BLK_K) {
        const bool hasNext = (k0 + BLK_K) < Kdim;
        const int  nxt     = cur ^ 1;

#if HAVE_ASYNC_LDS
        if (hasNext) {                       // DMA next tile into alternate buffer
            ASYNC_CP16(aPtr0 + (k0 + BLK_K), lA0[nxt]);
            ASYNC_CP16(aPtr1 + (k0 + BLK_K), lA1[nxt]);
            ASYNC_CP16(bPtr  + (k0 + BLK_K), lB[nxt]);
        }
#else
        uint4 na0, na1, nb;
        if (hasNext) {                       // issue next-tile loads early
            na0 = *(const uint4*)(aPtr0 + (k0 + BLK_K));
            na1 = *(const uint4*)(aPtr1 + (k0 + BLK_K));
            nb  = *(const uint4*)(bPtr  + (k0 + BLK_K));
        }
#endif
        if ((k0 + 2 * BLK_K) < Kdim) {       // gfx1250 global_prefetch at distance 2
            __builtin_prefetch(aPtr0 + (k0 + 2 * BLK_K), 0, 3);
            __builtin_prefetch(bPtr  + (k0 + 2 * BLK_K), 0, 3);
        }

        // ---- fragments per ISA 16-bit A/B lane layouts ----
        v16bf afr[2], bfr[2];
        #pragma unroll
        for (int mt = 0; mt < 2; ++mt) {
            const unsigned short* ap = &As[cur][(waveM * 32 + mt * 16 + l16) * LDA];
            union { unsigned int u[8]; v16bf v; } t;
            #pragma unroll
            for (int j = 0; j < 8; ++j) {
                int kk = (j < 4) ? (g * 8 + 2 * j) : (16 + g * 8 + 2 * (j - 4));
                t.u[j] = *(const unsigned int*)(ap + kk);
            }
            afr[mt] = t.v;
        }
        #pragma unroll
        for (int nt = 0; nt < 2; ++nt) {
            const unsigned short* bp = &Bs[cur][(waveN * 32 + nt * 16 + l16) * LDB];
            union { unsigned int u[8]; v16bf v; } t;
            #pragma unroll
            for (int j = 0; j < 8; ++j) {
                int kk = (j < 4) ? (g * 8 + 2 * j) : (16 + g * 8 + 2 * (j - 4));
                t.u[j] = *(const unsigned int*)(bp + kk);
            }
            bfr[nt] = t.v;
        }
        #pragma unroll
        for (int mt = 0; mt < 2; ++mt)
            #pragma unroll
            for (int nt = 0; nt < 2; ++nt)
                acc[mt][nt] = __builtin_amdgcn_wmma_f32_16x16x32_bf16(
                    false, afr[mt], false, bfr[nt], (short)0, acc[mt][nt], false, false);

#if HAVE_ASYNC_LDS
        if (hasNext) WAIT_ASYNC();           // own DMA writes landed in LDS
#else
        if (hasNext) {                       // stage into alternate buffer
            unsigned int* dp;
            dp = (unsigned int*)lA0[nxt]; dp[0] = na0.x; dp[1] = na0.y; dp[2] = na0.z; dp[3] = na0.w;
            dp = (unsigned int*)lA1[nxt]; dp[0] = na1.x; dp[1] = na1.y; dp[2] = na1.z; dp[3] = na1.w;
            dp = (unsigned int*)lB[nxt];  dp[0] = nb.x;  dp[1] = nb.y;  dp[2] = nb.z;  dp[3] = nb.w;
        }
#endif
        __syncthreads();                     // publish nxt, retire reads of cur
        if (hasNext) cur = nxt;
    }

    // ---- epilogue: bias, strided f32 and/or bf16 stores ----
    #pragma unroll
    for (int mt = 0; mt < 2; ++mt)
        #pragma unroll
        for (int nt = 0; nt < 2; ++nt) {
            int n = n0 + waveN * 32 + nt * 16 + l16;
            float bv = bias ? bias[n] : 0.0f;
            #pragma unroll
            for (int r = 0; r < 8; ++r) {
                int m = m0 + waveM * 32 + mt * 16 + r + 8 * g;
                float v = acc[mt][nt][r] + bv;
                if (Cf) Cf[(size_t)m * cm + (size_t)n * cn] = v;
                if (Cb) Cb[(size_t)m * bm + (size_t)n * bn] = f2bf(v);
            }
        }
}

// ---------------------------------------------------------------------------
// Depthwise causal conv (K=4) + SiLU. xz row-major [M][768], xi = cols 0..383
// ---------------------------------------------------------------------------
__global__ void conv_silu_kernel(const float* __restrict__ xz,
                                 const float* __restrict__ cw,
                                 const float* __restrict__ cb,
                                 float* __restrict__ xcf,
                                 unsigned short* __restrict__ xcb) {
    int m = blockIdx.x, d = threadIdx.x;
    int l = m & (Lsz - 1);
    float acc = cb[d];
    #pragma unroll
    for (int j = 0; j < 4; ++j) {
        int lp = l - 3 + j;
        if (lp >= 0) acc = fmaf(xz[(size_t)(m - 3 + j) * (2 * DIc) + d], cw[d * 4 + j], acc);
    }
    float s = acc / (1.0f + __expf(-acc));
    xcf[(size_t)m * DIc + d] = s;
    xcb[(size_t)m * DIc + d] = f2bf(s);
}

// ---------------------------------------------------------------------------
// dt = softplus(dbl[:, :12] @ dt_proj_w^T + dt_b)   (K=12 -> VALU)
// ---------------------------------------------------------------------------
__global__ void dt_kernel(const float* __restrict__ dbl,
                          const float* __restrict__ dtw,
                          const float* __restrict__ dtb,
                          float* __restrict__ dta) {
    __shared__ float s[Rr];
    int m = blockIdx.x, d = threadIdx.x;
    if (d < Rr) s[d] = dbl[(size_t)m * XPN + d];
    __syncthreads();
    float acc = dtb[d];
    #pragma unroll
    for (int r = 0; r < Rr; ++r) acc = fmaf(dtw[d * Rr + r], s[r], acc);
    float sp = (acc > 20.0f) ? acc : log1pf(__expf(acc));
    dta[(size_t)m * DIc + d] = sp;
}

// ---------------------------------------------------------------------------
// Selective scan: lane = (d,n) state. 16 lanes/channel, 2 ch per wave32,
// 16 channels per 256-thread block, 48 blocks. dA/dBx computed on the fly;
// y = sum_n(h*C) via 4-step shfl_xor tree within the 16-lane group.
// ---------------------------------------------------------------------------
__launch_bounds__(256)
__global__ void scan_kernel(const float* __restrict__ dta,
                            const float* __restrict__ dbl,
                            const float* __restrict__ xcf,
                            const float* __restrict__ A_log,
                            float* __restrict__ ys) {
    int bid  = blockIdx.x;                 // 0..47
    int b    = bid / (DIc / 16);
    int d0   = (bid % (DIc / 16)) * 16;
    int tid  = threadIdx.x;
    int lane = tid & 31;
    int wave = tid >> 5;
    int g    = lane >> 4;
    int n    = lane & 15;
    int d    = d0 + wave * 2 + g;

    float Adn = -__expf(A_log[d * Nst + n]);
    float h   = 0.0f;

    const float* dtp = dta + (size_t)b * Lsz * DIc + d;
    const float* xcp = xcf + (size_t)b * Lsz * DIc + d;
    const float* Bp  = dbl + (size_t)b * Lsz * XPN + Rr + n;
    const float* Cp  = dbl + (size_t)b * Lsz * XPN + Rr + Nst + n;
    float*       yp  = ys  + (size_t)b * Lsz * DIc + d;

    #pragma unroll 4
    for (int l = 0; l < Lsz; ++l) {
        float dt = dtp[(size_t)l * DIc];
        float xv = xcp[(size_t)l * DIc];
        float Bn = Bp[(size_t)l * XPN];
        float Cn = Cp[(size_t)l * XPN];
        float dA = __expf(dt * Adn);
        h = fmaf(dA, h, dt * Bn * xv);
        float y = h * Cn;
        y += __shfl_xor(y, 1, 32);
        y += __shfl_xor(y, 2, 32);
        y += __shfl_xor(y, 4, 32);
        y += __shfl_xor(y, 8, 32);
        if (n == 0) yp[(size_t)l * DIc] = y;
    }
}

// ---------------------------------------------------------------------------
// y = (ys + D*xc) * silu(z)  -> bf16 A for out_proj
// ---------------------------------------------------------------------------
__global__ void gate_kernel(const float* __restrict__ ys,
                            const float* __restrict__ xcf,
                            const float* __restrict__ xz,
                            const float* __restrict__ Dv,
                            unsigned short* __restrict__ yg) {
    int f = blockIdx.x * 256 + threadIdx.x;     // flat over [M][384]
    if (f >= Msz * DIc) return;
    int m = f / DIc, d = f % DIc;
    float z  = xz[(size_t)m * (2 * DIc) + DIc + d];
    float sz = z / (1.0f + __expf(-z));
    float y  = (ys[f] + Dv[d] * xcf[f]) * sz;
    yg[f] = f2bf(y);
}

// ---------------------------------------------------------------------------
// Per-channel batchnorm: xout channel-major [192][8192] -> (b,c,h,w)
// ---------------------------------------------------------------------------
__launch_bounds__(256)
__global__ void bn_kernel(const float* __restrict__ xout,
                          const float* __restrict__ gamma,
                          const float* __restrict__ beta,
                          float* __restrict__ out) {
    int o = blockIdx.x, t = threadIdx.x;
    const float* src = xout + (size_t)o * Msz;
    float s = 0.0f, s2 = 0.0f;
    for (int i = t; i < Msz; i += 256) { float v = src[i]; s += v; s2 += v * v; }
    __shared__ float sh[256], sh2[256];
    sh[t] = s; sh2[t] = s2; __syncthreads();
    for (int off = 128; off > 0; off >>= 1) {
        if (t < off) { sh[t] += sh[t + off]; sh2[t] += sh2[t + off]; }
        __syncthreads();
    }
    float mean = sh[0] * (1.0f / Msz);
    float var  = sh2[0] * (1.0f / Msz) - mean * mean;
    float inv  = rsqrtf(var + 1e-5f);
    float gm = gamma[o] * inv, bt = beta[o];
    for (int i = t; i < Msz; i += 256) {
        int b = i >> 12, l = i & (Lsz - 1);
        out[(size_t)b * DIMc * Lsz + (size_t)o * Lsz + l] = (src[i] - mean) * gm + bt;
    }
}

// ---------------------------------------------------------------------------
// Launch
// ---------------------------------------------------------------------------
extern "C" void kernel_launch(void* const* d_in, const int* in_sizes, int n_in,
                              void* d_out, int out_size, void* d_ws, size_t ws_size,
                              hipStream_t stream) {
    (void)in_sizes; (void)n_in; (void)out_size; (void)ws_size;
    const float* enc       = (const float*)d_in[0];
    const float* dec       = (const float*)d_in[1];
    const float* fuse_w    = (const float*)d_in[2];
    const float* fuse_b    = (const float*)d_in[3];
    const float* in_proj_w = (const float*)d_in[4];
    const float* in_proj_b = (const float*)d_in[5];
    const float* conv_w    = (const float*)d_in[6];
    const float* conv_b    = (const float*)d_in[7];
    const float* x_proj_w  = (const float*)d_in[8];
    const float* dt_proj_w = (const float*)d_in[9];
    const float* dt_proj_b = (const float*)d_in[10];
    const float* A_log     = (const float*)d_in[11];
    const float* Dv        = (const float*)d_in[12];
    const float* out_proj_w= (const float*)d_in[13];
    const float* out_proj_b= (const float*)d_in[14];
    const float* bn_gamma  = (const float*)d_in[15];
    const float* bn_beta   = (const float*)d_in[16];
    float* out = (float*)d_out;

    // bump allocator over workspace
    size_t off = 0;
    char* ws = (char*)d_ws;
    auto alloc = [&](size_t bytes) -> void* {
        void* p = ws + off;
        off += (bytes + 255) & ~(size_t)255;
        return p;
    };
    unsigned short* A1a  = (unsigned short*)alloc((size_t)Msz*2*DIMc*2);   // concat bf16 [M][384]
    unsigned short* fwb  = (unsigned short*)alloc((size_t)DIMc*2*DIMc*2);  // fuse W
    unsigned short* iwb  = (unsigned short*)alloc((size_t)2*DIc*DIMc*2);   // in_proj W
    unsigned short* xwb  = (unsigned short*)alloc((size_t)XPN*DIc*2);      // x_proj W padded
    unsigned short* owb  = (unsigned short*)alloc((size_t)DIMc*DIc*2);     // out_proj W
    unsigned short* A1   = (unsigned short*)alloc((size_t)Msz*DIMc*2);     // xfused bf16 [M][192]
    float*          xz   = (float*)alloc((size_t)Msz*2*DIc*4);             // [M][768]
    float*          xcf  = (float*)alloc((size_t)Msz*DIc*4);
    unsigned short* xcb  = (unsigned short*)alloc((size_t)Msz*DIc*2);
    float*          dbl  = (float*)alloc((size_t)Msz*XPN*4);               // [M][64]
    float*          dta  = (float*)alloc((size_t)Msz*DIc*4);
    float*          ysb  = (float*)alloc((size_t)Msz*DIc*4);
    unsigned short* yg   = (unsigned short*)alloc((size_t)Msz*DIc*2);
    float*          xout = (float*)alloc((size_t)DIMc*Msz*4);              // [192][8192]

    // 1. pack concat -> bf16 row-major, convert weights
    pack_fuse_A<<<Msz, 2*DIMc, 0, stream>>>(enc, dec, A1a);
    cvt_w<<<(DIMc*2*DIMc + 255)/256, 256, 0, stream>>>(fuse_w,     fwb, DIMc*2*DIMc, DIMc*2*DIMc);
    cvt_w<<<(2*DIc*DIMc  + 255)/256, 256, 0, stream>>>(in_proj_w,  iwb, 2*DIc*DIMc,  2*DIc*DIMc);
    cvt_w<<<(XPN*DIc     + 255)/256, 256, 0, stream>>>(x_proj_w,   xwb, (Rr+2*Nst)*DIc, XPN*DIc);
    cvt_w<<<(DIMc*DIc    + 255)/256, 256, 0, stream>>>(out_proj_w, owb, DIMc*DIc, DIMc*DIc);

    // 2. fuse GEMM: [8192 x 384] x [384 x 192] -> bf16 A1 (row-major [M][192])
    gemm_bf16_kernel<<<dim3(DIMc/BLK_N, Msz/BLK_M), 256, 0, stream>>>(
        A1a, 2*DIMc, fwb, 2*DIMc, fuse_b,
        (float*)nullptr, 0, 0, A1, DIMc, 1);

    // 3. in_proj GEMM: [8192 x 192] x [192 x 768] -> f32 xz [M][768]
    gemm_bf16_kernel<<<dim3(2*DIc/BLK_N, Msz/BLK_M), 256, 0, stream>>>(
        A1, DIMc, iwb, DIMc, in_proj_b,
        xz, 2*DIc, 1, (unsigned short*)nullptr, 0, 0);

    // 4. depthwise conv + silu -> xcf (f32) + xcb (bf16)
    conv_silu_kernel<<<Msz, DIc, 0, stream>>>(xz, conv_w, conv_b, xcf, xcb);

    // 5. x_proj GEMM: [8192 x 384] x [384 x 64pad] -> f32 dbl [M][64]
    gemm_bf16_kernel<<<dim3(XPN/BLK_N, Msz/BLK_M), 256, 0, stream>>>(
        xcb, DIc, xwb, DIc, (const float*)nullptr,
        dbl, XPN, 1, (unsigned short*)nullptr, 0, 0);

    // 6. dt projection + softplus
    dt_kernel<<<Msz, DIc, 0, stream>>>(dbl, dt_proj_w, dt_proj_b, dta);

    // 7. selective scan
    scan_kernel<<<Bz*(DIc/16), 256, 0, stream>>>(dta, dbl, xcf, A_log, ysb);

    // 8. gating -> bf16 yg
    gate_kernel<<<(Msz*DIc + 255)/256, 256, 0, stream>>>(ysb, xcf, xz, Dv, yg);

    // 9. out_proj GEMM -> xout channel-major [192][8192]
    gemm_bf16_kernel<<<dim3(DIMc/BLK_N, Msz/BLK_M), 256, 0, stream>>>(
        yg, DIc, owb, DIc, out_proj_b,
        xout, 1, Msz, (unsigned short*)nullptr, 0, 0);

    // 10. per-channel batchnorm -> (b, c, h, w) output
    bn_kernel<<<DIMc, 256, 0, stream>>>(xout, bn_gamma, bn_beta, out);
}
